// MultiHeadSelfAttention_50139448214192
// MI455X (gfx1250) — matrix operand order repaired and checked by reading, not compiled, not run on hardware
//
#include <hip/hip_runtime.h>

// ---------------------------------------------------------------------------
// Multi-head self-attention, f16 WMMA pipeline for gfx1250 (MI455X).
//   B=2, S=2048, D=1024, H=16, Dh=64.
// GEMMs: 128x128 block tile, LDS double-buffered (async-to-LDS when available),
// each of 8 waves computes a 32x64 sub-tile (8 WMMAs per K=32 step).
// ---------------------------------------------------------------------------

typedef __attribute__((ext_vector_type(16))) _Float16 v16h;
typedef __attribute__((ext_vector_type(8)))  _Float16 v8h;
typedef __attribute__((ext_vector_type(8)))  float    v8f;
typedef __attribute__((ext_vector_type(4)))  int      v4i;

#define B_   2
#define S_   2048
#define D_   1024
#define H_   16
#define DH_  64
#define M_   (B_ * S_)   // 4096 total rows

#define BM_  128
#define BN_  128
#define BK_  32

#if __has_builtin(__builtin_amdgcn_global_load_async_to_lds_b128) && \
    __has_builtin(__builtin_amdgcn_s_wait_asynccnt)
#define USE_ASYNC_LDS 1
typedef __attribute__((address_space(1))) v4i* gv4i_p;
typedef __attribute__((address_space(3))) v4i* lv4i_p;
#else
#define USE_ASYNC_LDS 0
#endif

// ---- copy 16 bytes global -> LDS (async path uses ASYNCcnt) ---------------
static __device__ __forceinline__ void cp_b128(const _Float16* g, _Float16* l) {
#if USE_ASYNC_LDS
  __builtin_amdgcn_global_load_async_to_lds_b128((gv4i_p)(g), (lv4i_p)(l), 0, 0);
#else
  *reinterpret_cast<v8h*>(l) = *reinterpret_cast<const v8h*>(g);
#endif
}

static __device__ __forceinline__ void cp_wait_barrier() {
#if USE_ASYNC_LDS
  __builtin_amdgcn_s_wait_asynccnt(0);
#endif
  __syncthreads();
}

// ---- WMMA fragment helpers (layouts per CDNA5 ISA §7.12.2, wave32) --------

// A fragment: 16x32 (MxK), 16-bit. lane<16: M=lane, K in {0..7}U{16..23};
// lane>=16: M=lane-16, K in {8..15}U{24..31}. Source row-major, leading dim ld.
static __device__ __forceinline__ v16h load_a_rowmajor(const _Float16* base, int ld, int lane) {
  const int row  = lane & 15;
  const int koff = (lane >> 4) << 3;               // 0 or 8
  const _Float16* p = base + row * ld + koff;
  v8h lo = *reinterpret_cast<const v8h*>(p);
  v8h hi = *reinterpret_cast<const v8h*>(p + 16);
  v16h a;
#pragma unroll
  for (int i = 0; i < 8; ++i) { a[i] = lo[i]; a[i + 8] = hi[i]; }
  return a;
}

// B fragment: 32x16 (KxN), 16-bit. lane<16: N=lane, K in {0..7}U{16..23};
// lane>=16: N=lane-16, K in {8..15}U{24..31}. Source K-major (column N contiguous over K, ld=ldk).
static __device__ __forceinline__ v16h load_b_kmajor(const _Float16* baseT, int ldk, int lane) {
  const int col  = lane & 15;
  const int koff = (lane >> 4) << 3;               // 0 or 8
  const _Float16* p = baseT + col * ldk + koff;
  v8h lo = *reinterpret_cast<const v8h*>(p);
  v8h hi = *reinterpret_cast<const v8h*>(p + 16);
  v16h b;
#pragma unroll
  for (int i = 0; i < 8; ++i) { b[i] = lo[i]; b[i + 8] = hi[i]; }
  return b;
}

static __device__ __forceinline__ v8f wmma_f16(v16h a, v16h b, v8f c) {
  return __builtin_amdgcn_wmma_f32_16x16x32_f16(false, a, false, b, (short)0, c, false, false);
}

// C/D fragment mapping: VGPR r holds (M = r + 8*(lane>=16), N = lane&15).

// ---- Conversion kernels ----------------------------------------------------

__global__ __launch_bounds__(256) void cvt_f32_to_f16_kernel(
    const float* __restrict__ src, _Float16* __restrict__ dst, int n) {
  int i = blockIdx.x * 256 + threadIdx.x;
  if (i < n) dst[i] = (_Float16)src[i];
}

// W (K x N row-major f32) -> Wt (N x K "K-major" f16): Wt[n*K + k] = W[k*N + n]
__global__ __launch_bounds__(256) void transpose_cvt_kernel(
    const float* __restrict__ W, _Float16* __restrict__ Wt) {
  int i = blockIdx.x * 256 + threadIdx.x;
  if (i >= D_ * D_) return;
  const int k = i >> 10;
  const int n = i & (D_ - 1);
  Wt[n * D_ + k] = (_Float16)W[i];
}

// ---- Block-tiled GEMM: Out = X(4096x1024) @ W + bias ----------------------
// mode 0: f16 out, (B,H,S,Dh) layout   (Q, K)
// mode 1: f16 out, (B,H,Dh,S) layout   (V transposed, ready as P@V B-operand)
// mode 2: f32 out, row-major           (final projection)

__global__ __launch_bounds__(256) void gemm_kernel(
    const _Float16* __restrict__ X, const _Float16* __restrict__ Wt,
    const float* __restrict__ bias, _Float16* __restrict__ Out16,
    float* __restrict__ Out32, int mode) {
  __shared__ _Float16 sA[2][BM_ * BK_];   // row-major within tile: [m][k]
  __shared__ _Float16 sB[2][BN_ * BK_];   // K-major within tile:  [n][k]

  const int tid  = threadIdx.x;
  const int lane = tid & 31;
  const int wave = tid >> 5;
  const int wm   = wave & 3;     // 4 waves down M: 32 rows each
  const int wn   = wave >> 2;    // 2 waves across N: 64 cols each

  const int m0b = (blockIdx.x >> 3) * BM_;   // M/BM = 32
  const int n0b = (blockIdx.x & 7) * BN_;    // N/BN = 8

  // prologue: stage K-step 0 into buffer 0 (512 x 16B chunks per tile)
#pragma unroll
  for (int t = 0; t < 2; ++t) {
    const int c = tid + t * 256;
    const int row = c >> 2;
    const int j   = (c & 3) << 3;
    cp_b128(X  + (size_t)(m0b + row) * D_ + j, &sA[0][row * BK_ + j]);
    cp_b128(Wt + (size_t)(n0b + row) * D_ + j, &sB[0][row * BK_ + j]);
  }
  cp_wait_barrier();

  v8f acc[2][4] = {};
  int buf = 0;
  for (int k0 = 0; k0 < D_; k0 += BK_) {
    const int nb = buf ^ 1;
    if (k0 + BK_ < D_) {
#pragma unroll
      for (int t = 0; t < 2; ++t) {
        const int c = tid + t * 256;
        const int row = c >> 2;
        const int j   = (c & 3) << 3;
        cp_b128(X  + (size_t)(m0b + row) * D_ + (k0 + BK_) + j, &sA[nb][row * BK_ + j]);
        cp_b128(Wt + (size_t)(n0b + row) * D_ + (k0 + BK_) + j, &sB[nb][row * BK_ + j]);
      }
    }

    // compute current buffer: 2 A-frags x 4 B-frags = 8 WMMAs
    const _Float16* sAb = sA[buf];
    const _Float16* sBb = sB[buf];
    v16h a[2];
#pragma unroll
    for (int mi = 0; mi < 2; ++mi)
      a[mi] = load_a_rowmajor(sAb + (wm * 32 + mi * 16) * BK_, BK_, lane);
#pragma unroll
    for (int nt = 0; nt < 4; ++nt) {
      v16h b = load_b_kmajor(sBb + (wn * 64 + nt * 16) * BK_, BK_, lane);
      acc[0][nt] = wmma_f16(a[0], b, acc[0][nt]);
      acc[1][nt] = wmma_f16(a[1], b, acc[1][nt]);
    }

    cp_wait_barrier();   // copies for next step done; all reads of buf done
    buf = nb;
  }

  // epilogue
  const int rbase = (lane >> 4) << 3;
  const int cn    = lane & 15;
#pragma unroll
  for (int mi = 0; mi < 2; ++mi) {
#pragma unroll
    for (int nt = 0; nt < 4; ++nt) {
      const int n  = n0b + wn * 64 + nt * 16 + cn;
      const float bb = bias[n];
      const int h = n >> 6, d = n & 63;
#pragma unroll
      for (int r = 0; r < 8; ++r) {
        const int m = m0b + wm * 32 + mi * 16 + rbase + r;
        const float val = acc[mi][nt][r] + bb;
        if (mode == 2) {
          Out32[(size_t)m * D_ + n] = val;
        } else {
          const int bidx = m >> 11;            // / S_
          const int s    = m & (S_ - 1);
          const long idx = (mode == 0)
              ? ((((long)(bidx * H_ + h)) * S_ + s) * DH_ + d)
              : ((((long)(bidx * H_ + h)) * DH_ + d) * S_ + s);
          Out16[idx] = (_Float16)val;
        }
      }
    }
  }
}

// ---- Attention: one block = (b, h, 16-query tile); 4 waves split keys -----
// LDS: sc[16][2048] f32 score strip (128 KB, inside 320 KB/WGP),
//      red[128] partials, stat[32] rowmax/invsum.

__global__ __launch_bounds__(128) void attn_kernel(
    const _Float16* __restrict__ Qh, const _Float16* __restrict__ Kh,
    const _Float16* __restrict__ Vt, _Float16* __restrict__ Ctx16) {
  extern __shared__ float smem[];
  float* sc   = smem;               // 16 * S_
  float* red  = smem + 16 * S_;     // 128
  float* stat = red + 128;          // 32 (rowmax @0, invsum @16)

  const int tid  = threadIdx.x;
  const int lane = tid & 31;
  const int wave = tid >> 5;

  const int qt = blockIdx.x & (S_ / 16 - 1);   // 0..127
  const int bh = blockIdx.x >> 7;              // 0..B*H-1
  const int m0 = qt * 16;
  const int b  = bh >> 4;
  const int h  = bh & (H_ - 1);

  const _Float16* Qb = Qh + (long)bh * S_ * DH_;
  const _Float16* Kb = Kh + (long)bh * S_ * DH_;
  const _Float16* Vb = Vt + (long)bh * DH_ * S_;

  const int rbase = (lane >> 4) << 3;
  const int cn    = lane & 15;

  // Q fragments for the whole tile: 16 x 64 = two 16x32 A-fragments.
  v16h aq0 = load_a_rowmajor(Qb + m0 * DH_ + 0,  DH_, lane);
  v16h aq1 = load_a_rowmajor(Qb + m0 * DH_ + 32, DH_, lane);

  const int tlo = wave * (S_ / 4);
  const int thi = tlo + (S_ / 4);

  // Phase 1: scores strip -> LDS (scaled by 1/sqrt(Dh) = 0.125)
  for (int t0 = tlo; t0 < thi; t0 += 16) {
    v8f c = {};
    v16h bk0 = load_b_kmajor(Kb + t0 * DH_ + 0,  DH_, lane);
    c = wmma_f16(aq0, bk0, c);
    v16h bk1 = load_b_kmajor(Kb + t0 * DH_ + 32, DH_, lane);
    c = wmma_f16(aq1, bk1, c);
#pragma unroll
    for (int r = 0; r < 8; ++r)
      sc[(rbase + r) * S_ + t0 + cn] = c[r] * 0.125f;
  }
  __syncthreads();

  // Phase 2: row softmax (8 threads per row)
  const int row = tid >> 3;
  const int sub = tid & 7;
  float mx = -3.0e38f;
  for (int t = sub; t < S_; t += 8) mx = fmaxf(mx, sc[row * S_ + t]);
  red[row * 8 + sub] = mx;
  __syncthreads();
  if (sub == 0) {
    float m = red[row * 8];
#pragma unroll
    for (int i = 1; i < 8; ++i) m = fmaxf(m, red[row * 8 + i]);
    stat[row] = m;
  }
  __syncthreads();
  const float rmax = stat[row];
  float sum = 0.f;
  for (int t = sub; t < S_; t += 8) {
    float e = __expf(sc[row * S_ + t] - rmax);
    sc[row * S_ + t] = e;
    sum += e;
  }
  red[row * 8 + sub] = sum;
  __syncthreads();
  if (sub == 0) {
    float s = 0.f;
#pragma unroll
    for (int i = 0; i < 8; ++i) s += red[row * 8 + i];
    stat[16 + row] = 1.0f / s;
  }
  __syncthreads();

  // Phase 3: partial ctx = P[tlo:thi] @ V[tlo:thi], per wave
  v8f acc[4] = {};
  for (int t0 = tlo; t0 < thi; t0 += 32) {
    v16h ap;
    const float* pr = sc + (lane & 15) * S_ + t0 + ((lane >> 4) << 3);
#pragma unroll
    for (int i = 0; i < 8; ++i) {
      ap[i]     = (_Float16)pr[i];
      ap[i + 8] = (_Float16)pr[16 + i];
    }
#pragma unroll
    for (int nt = 0; nt < 4; ++nt) {
      v16h bv = load_b_kmajor(Vb + (nt * 16) * S_ + t0, S_, lane);
      acc[nt] = wmma_f16(ap, bv, acc[nt]);
    }
  }
  __syncthreads();   // all waves done reading sc

  // Phase 4: scale by 1/rowsum, reduce the 4 wave partials via LDS
#pragma unroll
  for (int nt = 0; nt < 4; ++nt) {
#pragma unroll
    for (int r = 0; r < 8; ++r) {
      const int mrow = rbase + r;
      sc[(wave * 16 + mrow) * 64 + nt * 16 + cn] = acc[nt][r] * stat[16 + mrow];
    }
  }
  __syncthreads();
  for (int o = tid; o < 16 * 64; o += 128) {
    const int r = o >> 6, d = o & 63;
    const float v = sc[(0 * 16 + r) * 64 + d] + sc[(1 * 16 + r) * 64 + d] +
                    sc[(2 * 16 + r) * 64 + d] + sc[(3 * 16 + r) * 64 + d];
    Ctx16[((long)(b * S_ + m0 + r)) * D_ + h * DH_ + d] = (_Float16)v;
  }
}

// ---------------------------------------------------------------------------

extern "C" void kernel_launch(void* const* d_in, const int* in_sizes, int n_in,
                              void* d_out, int out_size, void* d_ws, size_t ws_size,
                              hipStream_t stream) {
  const float* x  = (const float*)d_in[0];
  const float* Wq = (const float*)d_in[1];
  const float* bq = (const float*)d_in[2];
  const float* Wk = (const float*)d_in[3];
  const float* bk = (const float*)d_in[4];
  const float* Wv = (const float*)d_in[5];
  const float* bv = (const float*)d_in[6];
  const float* Wo = (const float*)d_in[7];
  const float* bo = (const float*)d_in[8];
  float* out = (float*)d_out;

  char* ws = (char*)d_ws;
  _Float16* x16  = (_Float16*)ws; ws += (size_t)M_ * D_ * sizeof(_Float16);
  _Float16* Wqt  = (_Float16*)ws; ws += (size_t)D_ * D_ * sizeof(_Float16);
  _Float16* Wkt  = (_Float16*)ws; ws += (size_t)D_ * D_ * sizeof(_Float16);
  _Float16* Wvt  = (_Float16*)ws; ws += (size_t)D_ * D_ * sizeof(_Float16);
  _Float16* Wot  = (_Float16*)ws; ws += (size_t)D_ * D_ * sizeof(_Float16);
  _Float16* Qh   = (_Float16*)ws; ws += (size_t)M_ * D_ * sizeof(_Float16);
  _Float16* Kh   = (_Float16*)ws; ws += (size_t)M_ * D_ * sizeof(_Float16);
  _Float16* Vt   = (_Float16*)ws; ws += (size_t)M_ * D_ * sizeof(_Float16);
  _Float16* C16  = (_Float16*)ws; ws += (size_t)M_ * D_ * sizeof(_Float16);

  const int nx = M_ * D_;
  cvt_f32_to_f16_kernel<<<(nx + 255) / 256, 256, 0, stream>>>(x, x16, nx);

  const int nw = D_ * D_;
  const int wgrid = (nw + 255) / 256;
  transpose_cvt_kernel<<<wgrid, 256, 0, stream>>>(Wq, Wqt);
  transpose_cvt_kernel<<<wgrid, 256, 0, stream>>>(Wk, Wkt);
  transpose_cvt_kernel<<<wgrid, 256, 0, stream>>>(Wv, Wvt);
  transpose_cvt_kernel<<<wgrid, 256, 0, stream>>>(Wo, Wot);

  // (M/128)*(N/128) = 32*8 = 256 blocks, 256 threads (8 waves) each
  gemm_kernel<<<256, 256, 0, stream>>>(x16, Wqt, bq, Qh, nullptr, 0);
  gemm_kernel<<<256, 256, 0, stream>>>(x16, Wkt, bk, Kh, nullptr, 0);
  gemm_kernel<<<256, 256, 0, stream>>>(x16, Wvt, bv, Vt, nullptr, 1);

  const size_t shm = (size_t)(16 * S_ + 128 + 32) * sizeof(float);  // ~129 KB
  attn_kernel<<<B_ * H_ * (S_ / 16), 128, shm, stream>>>(Qh, Kh, Vt, C16);

  gemm_kernel<<<256, 256, 0, stream>>>(C16, Wot, bo, nullptr, out, 2);
}
